// SelfAttentionBlock_32332513804416
// MI455X (gfx1250) — matrix operand order repaired and checked by reading, not compiled, not run on hardware
//
#include <hip/hip_runtime.h>
#include <hip/hip_bf16.h>

typedef _Float16 half_t;
typedef __attribute__((ext_vector_type(16))) _Float16 v16h;
typedef __attribute__((ext_vector_type(8)))  _Float16 v8h;
typedef __attribute__((ext_vector_type(8)))  float    v8f;
typedef __attribute__((ext_vector_type(4)))  float    v4f;

static constexpr int Bn = 8;
static constexpr int Sn = 2048;
static constexpr int Dn = 768;
static constexpr int BS = Bn * Sn;          // 16384 rows
static constexpr float SCALE = 0.03608439182435161f;  // 1/sqrt(768)

// ---------------------------------------------------------------------------
// Kernel 0: transpose Wq/Wk/Wv (f32 [Din][Dout]) -> wt (f16 [mat][Dout][Din]).
// ---------------------------------------------------------------------------
__global__ __launch_bounds__(256) void wtrans_kernel(const float* __restrict__ Wq,
                                                     const float* __restrict__ Wk,
                                                     const float* __restrict__ Wv,
                                                     half_t* __restrict__ wt) {
  __shared__ half_t tile[32][33];
  const float* W = (blockIdx.z == 0) ? Wq : (blockIdx.z == 1) ? Wk : Wv;
  half_t* out = wt + (size_t)blockIdx.z * Dn * Dn;
  const int x = blockIdx.x * 32 + threadIdx.x;       // n (col of W)
  const int y0 = blockIdx.y * 32 + threadIdx.y;      // k (row of W)
  for (int i = 0; i < 32; i += 8)
    tile[threadIdx.y + i][threadIdx.x] = (half_t)W[(size_t)(y0 + i) * Dn + x];
  __syncthreads();
  const int k = blockIdx.y * 32 + threadIdx.x;
  for (int i = 0; i < 32; i += 8)
    out[(size_t)(blockIdx.x * 32 + threadIdx.y + i) * Dn + k] = tile[threadIdx.x][threadIdx.y + i];
}

// ---------------------------------------------------------------------------
// Kernel 1: LayerNorm (f32 in) -> xn (f16 out). One 256-thread block per row.
// ---------------------------------------------------------------------------
__global__ __launch_bounds__(256) void ln_kernel(const float* __restrict__ x,
                                                 const float* __restrict__ g,
                                                 const float* __restrict__ be,
                                                 half_t* __restrict__ xn) {
  __shared__ float red[256];
  const int row = blockIdx.x;
  const int t = threadIdx.x;
  const float* xr = x + (size_t)row * Dn;
  float v0 = xr[t], v1 = xr[t + 256], v2 = xr[t + 512];
  red[t] = v0 + v1 + v2;
  __syncthreads();
  for (int o = 128; o > 0; o >>= 1) { if (t < o) red[t] += red[t + o]; __syncthreads(); }
  const float mu = red[0] * (1.0f / Dn);
  __syncthreads();
  const float d0 = v0 - mu, d1 = v1 - mu, d2 = v2 - mu;
  red[t] = d0 * d0 + d1 * d1 + d2 * d2;
  __syncthreads();
  for (int o = 128; o > 0; o >>= 1) { if (t < o) red[t] += red[t + o]; __syncthreads(); }
  const float rstd = rsqrtf(red[0] * (1.0f / Dn) + 1e-5f);
  half_t* xo = xn + (size_t)row * Dn;
  xo[t]       = (half_t)(d0 * rstd * g[t]       + be[t]);
  xo[t + 256] = (half_t)(d1 * rstd * g[t + 256] + be[t + 256]);
  xo[t + 512] = (half_t)(d2 * rstd * g[t + 512] + be[t + 512]);
}

// ---------------------------------------------------------------------------
// Fragment helpers (all contiguous b128 loads)
// ---------------------------------------------------------------------------
__device__ __forceinline__ v16h load_a_f16(const half_t* arow, int k0, int hi) {
  v8h lo = *(const v8h*)(arow + k0 + hi * 16);
  v8h hc = *(const v8h*)(arow + k0 + hi * 16 + 8);
  v16h a;
  for (int i = 0; i < 8; ++i) { a[i] = lo[i]; a[8 + i] = hc[i]; }
  return a;
}
__device__ __forceinline__ v16h load_b_contig(const half_t* bp) {  // 16 contiguous halves
  v8h b0 = *(const v8h*)bp;
  v8h b1 = *(const v8h*)(bp + 8);
  v16h b;
  for (int i = 0; i < 8; ++i) { b[i] = b0[i]; b[8 + i] = b1[i]; }
  return b;
}
__device__ __forceinline__ v16h load_a_f32(const float* arow, int k0, int hi) {
  const float* p0 = arow + k0 + hi * 16;
  const float* p1 = arow + k0 + hi * 16 + 8;
  v4f x0 = *(const v4f*)(p0), x1 = *(const v4f*)(p0 + 4);
  v4f x2 = *(const v4f*)(p1), x3 = *(const v4f*)(p1 + 4);
  v16h a;
  for (int i = 0; i < 4; ++i) {
    a[i]      = (_Float16)x0[i];
    a[4 + i]  = (_Float16)x1[i];
    a[8 + i]  = (_Float16)x2[i];
    a[12 + i] = (_Float16)x3[i];
  }
  return a;
}
__device__ __forceinline__ v8f wmma_f16(v16h a, v16h b, v8f c) {
  return __builtin_amdgcn_wmma_f32_16x16x32_f16(false, a, false, b, (short)0, c, false, false);
}

// ---------------------------------------------------------------------------
// Kernel 2: QKV projections. grid=(1024,9), block=128 (4 waves).
// 16x64 strip per wave; double-buffered pipeline (next A+B loads in flight
// while current 4 WMMAs execute).
// ---------------------------------------------------------------------------
__global__ __launch_bounds__(128) void qkv_kernel(const half_t* __restrict__ xn,
    const half_t* __restrict__ wt,
    const float* __restrict__ bq, const float* __restrict__ bk, const float* __restrict__ bv,
    half_t* __restrict__ qh, half_t* __restrict__ kh, half_t* __restrict__ vt) {
  const int wave = threadIdx.x >> 5;
  const int lane = threadIdx.x & 31;
  const int nn = lane & 15;
  const int hi = lane >> 4;
  const int unit = blockIdx.y * 4 + wave;     // 0..35
  const int mat = unit / 12;                  // 0=Q 1=K 2=V
  const int n0 = (unit % 12) * 64;
  const int m0 = blockIdx.x * 16;

  const half_t* Wt = wt + (size_t)mat * Dn * Dn;
  const float* bias = (mat == 0) ? bq : (mat == 1) ? bk : bv;

  v8f c[4];
  for (int t = 0; t < 4; ++t) {
    const float bi = bias[n0 + t * 16 + nn];
    for (int r = 0; r < 8; ++r) c[t][r] = bi;
  }

  const half_t* arow = xn + (size_t)(m0 + nn) * Dn;
  const half_t* brow0 = Wt + (size_t)(n0 + nn) * Dn + hi * 16;        // tile stride 16*Dn

  v16h a = load_a_f16(arow, 0, hi);
  v16h b0 = load_b_contig(brow0);
  v16h b1 = load_b_contig(brow0 + (size_t)16 * Dn);
  v16h b2 = load_b_contig(brow0 + (size_t)32 * Dn);
  v16h b3 = load_b_contig(brow0 + (size_t)48 * Dn);

  for (int k0 = 32; k0 < Dn; k0 += 32) {
    v16h an  = load_a_f16(arow, k0, hi);
    v16h bn0 = load_b_contig(brow0 + k0);
    v16h bn1 = load_b_contig(brow0 + (size_t)16 * Dn + k0);
    v16h bn2 = load_b_contig(brow0 + (size_t)32 * Dn + k0);
    v16h bn3 = load_b_contig(brow0 + (size_t)48 * Dn + k0);
    c[0] = wmma_f16(a, b0, c[0]);
    c[1] = wmma_f16(a, b1, c[1]);
    c[2] = wmma_f16(a, b2, c[2]);
    c[3] = wmma_f16(a, b3, c[3]);
    a = an; b0 = bn0; b1 = bn1; b2 = bn2; b3 = bn3;
  }
  c[0] = wmma_f16(a, b0, c[0]);
  c[1] = wmma_f16(a, b1, c[1]);
  c[2] = wmma_f16(a, b2, c[2]);
  c[3] = wmma_f16(a, b3, c[3]);

  if (mat < 2) {
    half_t* out = (mat == 0) ? qh : kh;
    for (int t = 0; t < 4; ++t) {
      half_t* orow = out + (size_t)(m0 + hi * 8) * Dn + n0 + t * 16 + nn;
      for (int r = 0; r < 8; ++r) orow[(size_t)r * Dn] = (half_t)c[t][r];
    }
  } else {
    const int b  = m0 >> 11;                  // batch
    const int s0 = (m0 & 2047) + hi * 8;      // seq position of first row
    for (int t = 0; t < 4; ++t) {
      v8h pk;
      for (int r = 0; r < 8; ++r) pk[r] = (half_t)c[t][r];
      *(v8h*)(vt + ((size_t)b * Dn + n0 + t * 16 + nn) * Sn + s0) = pk;
    }
  }
}

// ---------------------------------------------------------------------------
// Kernel 3: scores = Q.K^T / sqrt(D) into probs region (raw).
// grid=(128,8,8), block=128. 16x64 strip, double-buffered pipeline.
// ---------------------------------------------------------------------------
__global__ __launch_bounds__(128) void scores_kernel(const half_t* __restrict__ qh,
                                                     const half_t* __restrict__ kh,
                                                     float* __restrict__ scores) {
  const int wave = threadIdx.x >> 5;
  const int lane = threadIdx.x & 31;
  const int nn = lane & 15;
  const int hi = lane >> 4;
  const int bz = blockIdx.z;
  const int q0 = blockIdx.x * 16;
  const int kg = (blockIdx.y * 4 + wave) * 64;

  const half_t* arow  = qh + ((size_t)bz * Sn + q0 + nn) * Dn;
  const half_t* brow0 = kh + ((size_t)bz * Sn + kg + nn) * Dn + hi * 16;  // tile stride 16*Dn

  v8f c[4];
  for (int t = 0; t < 4; ++t)
    for (int r = 0; r < 8; ++r) c[t][r] = 0.0f;

  v16h a = load_a_f16(arow, 0, hi);
  v16h b0 = load_b_contig(brow0);
  v16h b1 = load_b_contig(brow0 + (size_t)16 * Dn);
  v16h b2 = load_b_contig(brow0 + (size_t)32 * Dn);
  v16h b3 = load_b_contig(brow0 + (size_t)48 * Dn);

  for (int d0 = 32; d0 < Dn; d0 += 32) {
    v16h an  = load_a_f16(arow, d0, hi);
    v16h bn0 = load_b_contig(brow0 + d0);
    v16h bn1 = load_b_contig(brow0 + (size_t)16 * Dn + d0);
    v16h bn2 = load_b_contig(brow0 + (size_t)32 * Dn + d0);
    v16h bn3 = load_b_contig(brow0 + (size_t)48 * Dn + d0);
    c[0] = wmma_f16(a, b0, c[0]);
    c[1] = wmma_f16(a, b1, c[1]);
    c[2] = wmma_f16(a, b2, c[2]);
    c[3] = wmma_f16(a, b3, c[3]);
    a = an; b0 = bn0; b1 = bn1; b2 = bn2; b3 = bn3;
  }
  c[0] = wmma_f16(a, b0, c[0]);
  c[1] = wmma_f16(a, b1, c[1]);
  c[2] = wmma_f16(a, b2, c[2]);
  c[3] = wmma_f16(a, b3, c[3]);

  for (int t = 0; t < 4; ++t) {
    float* orow = scores + (size_t)bz * Sn * Sn + (size_t)(q0 + hi * 8) * Sn + kg + t * 16 + nn;
    for (int r = 0; r < 8; ++r) orow[(size_t)r * Sn] = c[t][r] * SCALE;
  }
}

// ---------------------------------------------------------------------------
// Kernel 4: per key-column (across queries) online max + sum(exp).
// ---------------------------------------------------------------------------
__global__ __launch_bounds__(256) void colstats_kernel(const float* __restrict__ probs,
                                                       float* __restrict__ mcol,
                                                       float* __restrict__ rinv) {
  const int col = blockIdx.x * 256 + threadIdx.x;   // 0..16383
  const int b = col >> 11;
  const int k = col & 2047;
  const float* p = probs + (size_t)b * Sn * Sn + k;
  float m = -3.4e38f, sum = 0.0f;
  for (int q = 0; q < Sn; ++q) {
    const float s = p[(size_t)q * Sn];
    const float mn = fmaxf(m, s);
    sum = sum * __expf(m - mn) + __expf(s - mn);
    m = mn;
  }
  mcol[col] = m;
  rinv[col] = 1.0f / sum;
}

// ---------------------------------------------------------------------------
// Kernel 5: in-place normalize probs = exp(s - m[col]) * rinv[col]. float4.
// ---------------------------------------------------------------------------
__global__ __launch_bounds__(256) void norm_kernel(float* __restrict__ probs,
                                                   const float* __restrict__ mcol,
                                                   const float* __restrict__ rinv) {
  const size_t idx4 = ((size_t)blockIdx.x * 256 + threadIdx.x) * 4;
  const int b  = (int)(idx4 >> 22);          // / (2048*2048)
  const int k0 = (int)(idx4 & 2047);
  const int cb = (b << 11) + k0;
  float4 s = *(const float4*)(probs + idx4);
  float4 o;
  o.x = __expf(s.x - mcol[cb + 0]) * rinv[cb + 0];
  o.y = __expf(s.y - mcol[cb + 1]) * rinv[cb + 1];
  o.z = __expf(s.z - mcol[cb + 2]) * rinv[cb + 2];
  o.w = __expf(s.w - mcol[cb + 3]) * rinv[cb + 3];
  *(float4*)(probs + idx4) = o;
}

// ---------------------------------------------------------------------------
// Kernel 6: out = P.V. grid=(128,3,8), block=128. 16x64 strip, pipelined.
// P converted f32->f16 per fragment (L2-resident), V read from transposed vt.
// ---------------------------------------------------------------------------
__global__ __launch_bounds__(128) void out_kernel(const float* __restrict__ probs,
                                                  const half_t* __restrict__ vt,
                                                  float* __restrict__ out) {
  const int wave = threadIdx.x >> 5;
  const int lane = threadIdx.x & 31;
  const int nn = lane & 15;
  const int hi = lane >> 4;
  const int bz = blockIdx.z;
  const int q0 = blockIdx.x * 16;
  const int n0 = (blockIdx.y * 4 + wave) * 64;

  const float*  arow  = probs + (size_t)bz * Sn * Sn + (size_t)(q0 + nn) * Sn;
  const half_t* brow0 = vt + ((size_t)bz * Dn + n0 + nn) * Sn + hi * 16;  // tile stride 16*Sn

  v8f c[4];
  for (int t = 0; t < 4; ++t)
    for (int r = 0; r < 8; ++r) c[t][r] = 0.0f;

  v16h a = load_a_f32(arow, 0, hi);
  v16h b0 = load_b_contig(brow0);
  v16h b1 = load_b_contig(brow0 + (size_t)16 * Sn);
  v16h b2 = load_b_contig(brow0 + (size_t)32 * Sn);
  v16h b3 = load_b_contig(brow0 + (size_t)48 * Sn);

  for (int k0 = 32; k0 < Sn; k0 += 32) {
    v16h an  = load_a_f32(arow, k0, hi);
    v16h bn0 = load_b_contig(brow0 + k0);
    v16h bn1 = load_b_contig(brow0 + (size_t)16 * Sn + k0);
    v16h bn2 = load_b_contig(brow0 + (size_t)32 * Sn + k0);
    v16h bn3 = load_b_contig(brow0 + (size_t)48 * Sn + k0);
    c[0] = wmma_f16(a, b0, c[0]);
    c[1] = wmma_f16(a, b1, c[1]);
    c[2] = wmma_f16(a, b2, c[2]);
    c[3] = wmma_f16(a, b3, c[3]);
    a = an; b0 = bn0; b1 = bn1; b2 = bn2; b3 = bn3;
  }
  c[0] = wmma_f16(a, b0, c[0]);
  c[1] = wmma_f16(a, b1, c[1]);
  c[2] = wmma_f16(a, b2, c[2]);
  c[3] = wmma_f16(a, b3, c[3]);

  for (int t = 0; t < 4; ++t) {
    float* orow = out + (size_t)bz * Sn * Dn + (size_t)(q0 + hi * 8) * Dn + n0 + t * 16 + nn;
    for (int r = 0; r < 8; ++r) orow[(size_t)r * Dn] = c[t][r];
  }
}

// ---------------------------------------------------------------------------
extern "C" void kernel_launch(void* const* d_in, const int* in_sizes, int n_in,
                              void* d_out, int out_size, void* d_ws, size_t ws_size,
                              hipStream_t stream) {
  const float* x  = (const float*)d_in[0];
  const float* g  = (const float*)d_in[1];
  const float* be = (const float*)d_in[2];
  const float* Wq = (const float*)d_in[3];
  const float* bq = (const float*)d_in[4];
  const float* Wk = (const float*)d_in[5];
  const float* bk = (const float*)d_in[6];
  const float* Wv = (const float*)d_in[7];
  const float* bv = (const float*)d_in[8];

  float* out   = (float*)d_out;
  float* probs = out + (size_t)Bn * Sn * Dn;          // second tuple element

  // Workspace layout (~104.3 MB)
  char* ws = (char*)d_ws;
  const size_t XNB = (size_t)BS * Dn * sizeof(half_t);     // 25,165,824 each
  const size_t WTB = (size_t)3 * Dn * Dn * sizeof(half_t); // 3,538,944
  half_t* xn = (half_t*)(ws);
  half_t* qh = (half_t*)(ws + XNB);
  half_t* kh = (half_t*)(ws + 2 * XNB);
  half_t* vt = (half_t*)(ws + 3 * XNB);
  half_t* wt = (half_t*)(ws + 4 * XNB);
  float* mcol = (float*)(ws + 4 * XNB + WTB);
  float* rinv = mcol + (size_t)Bn * Sn;

  wtrans_kernel<<<dim3(Dn / 32, Dn / 32, 3), dim3(32, 8), 0, stream>>>(Wq, Wk, Wv, wt);
  ln_kernel<<<BS, 256, 0, stream>>>(x, g, be, xn);
  qkv_kernel<<<dim3(BS / 16, 9), 128, 0, stream>>>(xn, wt, bq, bk, bv, qh, kh, vt);
  scores_kernel<<<dim3(Sn / 16, 8, Bn), 128, 0, stream>>>(qh, kh, probs);
  colstats_kernel<<<(Bn * Sn) / 256, 256, 0, stream>>>(probs, mcol, rinv);
  norm_kernel<<<(int)(((size_t)Bn * Sn * Sn) / 4 / 256), 256, 0, stream>>>(probs, mcol, rinv);
  out_kernel<<<dim3(Sn / 16, 3, Bn), 128, 0, stream>>>(probs, vt, out);
}
